// TreeLSTM_31980326486848
// MI455X (gfx1250) — compile-verified, matrix operand order checked by reading, not run on hardware
//
#include <hip/hip_runtime.h>
#include <hip/hip_bf16.h>
#include <stdint.h>

typedef unsigned short u16;

#define NNODES 255
#define SEQ_W  768
#define HDIM   2304
#define H2DIM  4608
#define H3DIM  6912
#define INEPS  1e-5f

typedef __bf16 v16bf __attribute__((ext_vector_type(16)));
typedef float  v8f   __attribute__((ext_vector_type(8)));

union FragBF { v16bf v; uint4 q[2]; };

__device__ __forceinline__ float sigm(float x) { return 1.0f / (1.0f + expf(-x)); }

__device__ __forceinline__ u16 to_bf16(float f) {
  unsigned int u = __float_as_uint(f);
  u += 0x7fffu + ((u >> 16) & 1u);
  return (u16)(u >> 16);
}

// ---------------------------------------------------------------------------
// f32 -> bf16 weight conversion (once per call; weights then live in L2)
// ---------------------------------------------------------------------------
__global__ void cvt_bf16_kernel(const float* __restrict__ src, u16* __restrict__ dst, long n) {
  long stride = (long)gridDim.x * blockDim.x;
  for (long i = (long)blockIdx.x * blockDim.x + threadIdx.x; i < n; i += stride)
    dst[i] = to_bf16(src[i]);
}

// ---------------------------------------------------------------------------
// WMMA bf16 GEMM:  C(Mpad x N) = act( A(Mpad x K) * W(N x K)^T + bias )
// A, W row-major bf16 (K contiguous). One block = 16 rows x 256 cols.
// 8 waves/block; each wave owns two 16-wide N tiles, reusing one A fragment.
// ---------------------------------------------------------------------------
__global__ void gemm_bf16_nt(const u16* __restrict__ A, const u16* __restrict__ W,
                             const float* __restrict__ bias, float* __restrict__ C,
                             int N, int K, int act) {
  const int lane = threadIdx.x & 31;
  const int wave = threadIdx.x >> 5;
  const int m16  = lane & 15;
  const int half = lane >> 4;
  const int tileM = blockIdx.y << 4;
  const int nBase = blockIdx.x * 256 + wave * 32;
  const int koff  = half * 8;

  const u16* arow  = A + (size_t)(tileM + m16) * K;
  const u16* wrow0 = W + (size_t)(nBase + m16) * K;
  const u16* wrow1 = wrow0 + (size_t)16 * K;

  v8f acc0 = {0.f,0.f,0.f,0.f,0.f,0.f,0.f,0.f};
  v8f acc1 = {0.f,0.f,0.f,0.f,0.f,0.f,0.f,0.f};

  for (int k0 = 0; k0 < K; k0 += 32) {
    FragBF a, b0, b1;
    a.q[0]  = *(const uint4*)(arow  + k0 + koff);
    a.q[1]  = *(const uint4*)(arow  + k0 + 16 + koff);
    b0.q[0] = *(const uint4*)(wrow0 + k0 + koff);
    b0.q[1] = *(const uint4*)(wrow0 + k0 + 16 + koff);
    b1.q[0] = *(const uint4*)(wrow1 + k0 + koff);
    b1.q[1] = *(const uint4*)(wrow1 + k0 + 16 + koff);
    __builtin_prefetch(wrow0 + k0 + 256, 0, 1);
    __builtin_prefetch(wrow1 + k0 + 256, 0, 1);
    acc0 = __builtin_amdgcn_wmma_f32_16x16x32_bf16(false, a.v, false, b0.v, (short)0, acc0, false, false);
    acc1 = __builtin_amdgcn_wmma_f32_16x16x32_bf16(false, a.v, false, b1.v, (short)0, acc1, false, false);
  }

  // C/D layout: N = lane%16, M = vgpr + 8*(lane/16)
  const int mBase = tileM + (half << 3);
  const int n0 = nBase + m16;
  for (int r = 0; r < 8; ++r) {
    float v0 = acc0[r], v1 = acc1[r];
    if (act) {
      v0 = sigm(v0 + bias[n0]);
      v1 = sigm(v1 + bias[n0 + 16]);
    }
    C[(size_t)(mBase + r) * N + n0]      = v0;
    C[(size_t)(mBase + r) * N + n0 + 16] = v1;
  }
}

// ---------------------------------------------------------------------------
// h_cat assembly (f32 h -> bf16 h_cat, zero-padded to npad rows)
// ---------------------------------------------------------------------------
__global__ void prep_up_kernel(const float* __restrict__ h, u16* __restrict__ hcat,
                               int lo, int n, int npad) {
  long total = (long)npad * H2DIM;
  long stride = (long)gridDim.x * blockDim.x;
  for (long idx = (long)blockIdx.x * blockDim.x + threadIdx.x; idx < total; idx += stride) {
    int row = (int)(idx / H2DIM);
    int col = (int)(idx % H2DIM);
    float v = 0.f;
    if (row < n) {
      int child = 2 * (lo + row) + 1 + (col >= HDIM ? 1 : 0);
      int cc = col < HDIM ? col : col - HDIM;
      v = h[(size_t)child * HDIM + cc];
    }
    hcat[idx] = to_bf16(v);
  }
}

__global__ void prep_down_kernel(const float* __restrict__ h, u16* __restrict__ hcat,
                                 int lo, int n, int npad) {
  long total = (long)npad * H2DIM;
  long stride = (long)gridDim.x * blockDim.x;
  for (long idx = (long)blockIdx.x * blockDim.x + threadIdx.x; idx < total; idx += stride) {
    int row = (int)(idx / H2DIM);
    int col = (int)(idx % H2DIM);
    float v = 0.f;
    if (row < n) {
      int parent = (lo + row - 1) >> 1;
      int cc = col < HDIM ? col : col - HDIM;
      v = h[(size_t)parent * HDIM + cc];
    }
    hcat[idx] = to_bf16(v);
  }
}

// ---------------------------------------------------------------------------
// reduce + msg_norm + apply_node, one block (256 threads) per tree node
// ---------------------------------------------------------------------------
__device__ float block_sum(float v, float* red) {
  int t = threadIdx.x;
  red[t] = v;
  __syncthreads();
  for (int s = 128; s > 0; s >>= 1) {
    if (t < s) red[t] += red[t + s];
    __syncthreads();
  }
  float r = red[0];
  __syncthreads();
  return r;
}

__global__ void reduce_apply_kernel(float* __restrict__ iou, float* __restrict__ h,
                                    float* __restrict__ c,
                                    const float* __restrict__ msg,   // n x 6912
                                    const float* __restrict__ f,     // n x 4608 (sigmoided)
                                    const float* __restrict__ b_iou,
                                    const float* __restrict__ s_iou_p,
                                    const float* __restrict__ s_c_p,
                                    int lo, int down) {
  __shared__ float red[256];
  __shared__ float csum[HDIM];
  const int i = blockIdx.x;
  const int node = lo + i;
  int c0, c1;
  if (down) { int p = (node - 1) >> 1; c0 = p; c1 = p; }
  else      { c0 = 2 * node + 1; c1 = c0 + 1; }
  const int tid = threadIdx.x;
  const float* mrow = msg + (size_t)i * H3DIM;
  const float* frow = f + (size_t)i * H2DIM;
  float* iourow = iou + (size_t)node * H3DIM;

  float sm = 0.f, sfeat = 0.f;
  for (int j = tid; j < H3DIM; j += 256) {
    float m = mrow[j];   sm    += m * m;
    float t = iourow[j]; sfeat += t * t;
  }
  float scs = 0.f, scf = 0.f;
  for (int j = tid; j < HDIM; j += 256) {
    float ca = c[(size_t)c0 * HDIM + j];
    float cb = c[(size_t)c1 * HDIM + j];
    float cs = frow[j] * ca + frow[HDIM + j] * cb;
    csum[j] = cs;
    scs += cs * cs;
    scf += ca * ca;
  }
  float nm  = sqrtf(block_sum(sm, red));
  float nf  = sqrtf(block_sum(sfeat, red));
  float ncs = sqrtf(block_sum(scs, red));
  float ncf = sqrtf(block_sum(scf, red));
  const float km = s_iou_p[0] * nf  / fmaxf(nm,  1e-12f);
  const float kc = s_c_p[0]   * ncf / fmaxf(ncs, 1e-12f);

  for (int j = tid; j < HDIM; j += 256) {
    float iN = iourow[j]            + km * mrow[j];
    float oN = iourow[HDIM + j]     + km * mrow[HDIM + j];
    float uN = iourow[2 * HDIM + j] + km * mrow[2 * HDIM + j];
    iourow[j] = iN; iourow[HDIM + j] = oN; iourow[2 * HDIM + j] = uN;
    float cnode = c[(size_t)c0 * HDIM + j] + kc * csum[j];
    float cn = sigm(iN + b_iou[j]) * tanhf(uN + b_iou[2 * HDIM + j]) + cnode;
    float hn = sigm(oN + b_iou[HDIM + j]) * tanhf(cn);
    c[(size_t)node * HDIM + j] = cn;
    h[(size_t)node * HDIM + j] = hn;
  }
}

__global__ void apply_node_kernel(const float* __restrict__ iou, const float* __restrict__ c_src,
                                  float* __restrict__ h, float* __restrict__ c,
                                  const float* __restrict__ b_iou, int lo, long total) {
  long stride = (long)gridDim.x * blockDim.x;
  for (long idx = (long)blockIdx.x * blockDim.x + threadIdx.x; idx < total; idx += stride) {
    int node = lo + (int)(idx / HDIM);
    int j = (int)(idx % HDIM);
    const float* iourow = iou + (size_t)node * H3DIM;
    float cn = sigm(iourow[j] + b_iou[j]) * tanhf(iourow[2 * HDIM + j] + b_iou[2 * HDIM + j])
             + c_src[(size_t)node * HDIM + j];
    c[(size_t)node * HDIM + j] = cn;
    h[(size_t)node * HDIM + j] = sigm(iourow[HDIM + j] + b_iou[HDIM + j]) * tanhf(cn);
  }
}

__global__ void relu_kernel(float* __restrict__ p, long n) {
  long stride = (long)gridDim.x * blockDim.x;
  for (long i = (long)blockIdx.x * blockDim.x + threadIdx.x; i < n; i += stride)
    p[i] = fmaxf(p[i], 0.f);
}

// ---------------------------------------------------------------------------
// In-LDS conv helpers (encoder/decoder), 256-thread workgroup, per-node
// ---------------------------------------------------------------------------
__device__ void conv3_lds(const float* in, int Cin, int W, const float* wgt, int o,
                          float* out, const float* resid) {
  for (int idx = threadIdx.x; idx < o * W; idx += blockDim.x) {
    int oc = idx / W, x = idx % W;
    float s = 0.f;
    const float* wr = wgt + (size_t)oc * Cin * 3;
    for (int ic = 0; ic < Cin; ++ic) {
      const float* row = in + ic * W;
      float l = (x > 0)     ? row[x - 1] : 0.f;
      float m = row[x];
      float r = (x < W - 1) ? row[x + 1] : 0.f;
      s += wr[ic * 3 + 0] * l + wr[ic * 3 + 1] * m + wr[ic * 3 + 2] * r;
    }
    if (resid) s += resid[idx];
    out[idx] = s;
  }
  __syncthreads();
}

__device__ void inorm_rows(float* buf, int C, int W) {
  for (int ch = threadIdx.x; ch < C; ch += blockDim.x) {
    float* row = buf + ch * W;
    float m = 0.f;
    for (int j = 0; j < W; ++j) m += row[j];
    m /= (float)W;
    float v = 0.f;
    for (int j = 0; j < W; ++j) { float d = row[j] - m; v += d * d; }
    v /= (float)W;
    float inv = rsqrtf(v + INEPS);
    for (int j = 0; j < W; ++j) row[j] = (row[j] - m) * inv;
  }
  __syncthreads();
}

__device__ void relu_buf(float* buf, int n) {
  for (int i = threadIdx.x; i < n; i += blockDim.x) buf[i] = fmaxf(buf[i], 0.f);
  __syncthreads();
}

__device__ void res_stack_lds(const float* in, int Cin, int W, const float* const* w3,
                              int o, float* out) {
  conv3_lds(in, Cin, W, w3[0], o, out, nullptr);
  inorm_rows(out, o, W);
  conv3_lds(out, o, W, w3[1], o, out + (size_t)o * W, out);
  inorm_rows(out + (size_t)o * W, o, W);
  conv3_lds(out + (size_t)o * W, o, W, w3[2], o, out + (size_t)2 * o * W, out + (size_t)o * W);
  inorm_rows(out + (size_t)2 * o * W, o, W);
  relu_buf(out, 3 * o * W);
}

// ---------------------------------------------------------------------------
// Encoder: one workgroup per node. bufIn 13824 f32, bufOut 27648 f32 (dyn LDS)
// ---------------------------------------------------------------------------
struct EncArgs {
  const float* w[12];
  const float* out_w;
  const float* out_b;
  const float* x;
  float* iou;
};

__global__ void encoder_kernel(EncArgs a) {
  extern __shared__ float esm[];
  float* bufIn  = esm;          // 13824 floats
  float* bufOut = esm + 13824;  // 27648 floats
  const int node = blockIdx.x;
  for (int j = threadIdx.x; j < SEQ_W; j += blockDim.x)
    bufIn[j] = a.x[(size_t)node * SEQ_W + j];
  __syncthreads();

  int Cin = 1, W = SEQ_W;
  const int OCH[4] = {8, 16, 32, 96};
  for (int b = 0; b < 4; ++b) {
    int o = OCH[b];
    const float* w3[3] = {a.w[b * 3 + 0], a.w[b * 3 + 1], a.w[b * 3 + 2]};
    res_stack_lds(bufIn, Cin, W, w3, o, bufOut);
    // maxpool (1,2) -> bufIn
    int C2 = 3 * o, W2 = W / 2;
    for (int idx = threadIdx.x; idx < C2 * W2; idx += blockDim.x) {
      int ch = idx / W2, x = idx % W2;
      float u = bufOut[(size_t)ch * W + 2 * x];
      float v = bufOut[(size_t)ch * W + 2 * x + 1];
      bufIn[idx] = fmaxf(u, v);
    }
    __syncthreads();
    inorm_rows(bufIn, C2, W2);
    relu_buf(bufIn, C2 * W2);
    Cin = C2; W = W2;
  }
  // 1x1 conv 288 -> 144, write iou (ch-major, 144 x 48)
  for (int idx = threadIdx.x; idx < 144 * 48; idx += blockDim.x) {
    int ch = idx / 48, x = idx % 48;
    float s = a.out_b[ch];
    const float* wr = a.out_w + (size_t)ch * 288;
    for (int k = 0; k < 288; ++k) s += wr[k] * bufIn[(size_t)k * 48 + x];
    a.iou[(size_t)node * H3DIM + idx] = s;
  }
}

// ---------------------------------------------------------------------------
// Decoder: one workgroup per node; bicubic upsample + res stacks in LDS
// ---------------------------------------------------------------------------
struct DecArgs {
  const float* w[12];
  const float* out_w;
  const float* out_b;
  const float* h;
  float* out;
};

__device__ void upsample2(const float* in, int C, int W, float* out) {
  const int W2 = 2 * W;
  const float scale = (float)(W - 1) / (float)(W2 - 1);
  const float A = -0.75f;
  for (int idx = threadIdx.x; idx < C * W2; idx += blockDim.x) {
    int ch = idx / W2, o = idx % W2;
    float src = (float)o * scale;
    int i0 = (int)floorf(src);
    float acc = 0.f;
    for (int t = -1; t <= 2; ++t) {
      int ix = i0 + t;
      float d = fabsf((float)ix - src);
      float wgt;
      if (d <= 1.f)      wgt = ((A + 2.f) * d - (A + 3.f)) * d * d + 1.f;
      else if (d < 2.f)  wgt = ((A * d - 5.f * A) * d + 8.f * A) * d - 4.f * A;
      else               wgt = 0.f;
      int ic = ix < 0 ? 0 : (ix > W - 1 ? W - 1 : ix);
      acc += in[(size_t)ch * W + ic] * wgt;
    }
    out[idx] = acc;
  }
  __syncthreads();
}

__global__ void decoder_kernel(DecArgs a) {
  __shared__ float dIn[6144];
  __shared__ float dOut[3072];
  const int node = blockIdx.x;
  for (int j = threadIdx.x; j < 144 * 16; j += blockDim.x)
    dOut[j] = a.h[(size_t)node * HDIM + j];
  __syncthreads();

  int C = 144, W = 16;
  const int OCH[4] = {32, 16, 8, 1};
  for (int b = 0; b < 4; ++b) {
    int o = OCH[b];
    upsample2(dOut, C, W, dIn);
    W *= 2;
    const float* w3[3] = {a.w[b * 3 + 0], a.w[b * 3 + 1], a.w[b * 3 + 2]};
    res_stack_lds(dIn, C, W, w3, o, dOut);
    C = 3 * o;
    inorm_rows(dOut, C, W);
  }
  // final 1x1 conv 3 -> 1, W == 256
  for (int x = threadIdx.x; x < 256; x += blockDim.x) {
    float s = a.out_b[0];
    for (int ch = 0; ch < 3; ++ch) s += a.out_w[ch] * dOut[(size_t)ch * 256 + x];
    a.out[(size_t)node * 256 + x] = s;
  }
}

// ---------------------------------------------------------------------------
// Host orchestration
// ---------------------------------------------------------------------------
static inline int gblocks(long total) {
  long b = (total + 255) / 256;
  return (int)(b > 16384 ? 16384 : (b < 1 ? 1 : b));
}

extern "C" void kernel_launch(void* const* d_in, const int* in_sizes, int n_in,
                              void* d_out, int out_size, void* d_ws, size_t ws_size,
                              hipStream_t stream) {
  (void)in_sizes; (void)n_in; (void)out_size; (void)ws_size;
  const float* x      = (const float*)d_in[0];
  const float* c_init = (const float*)d_in[2];
  const float* Wf     = (const float*)d_in[31];
  const float* bf     = (const float*)d_in[32];
  const float* Wiou   = (const float*)d_in[33];
  const float* b_iou  = (const float*)d_in[34];
  const float* s_iou  = (const float*)d_in[35];
  const float* s_c    = (const float*)d_in[36];

  // workspace layout (all 256B aligned sizes)
  char* ws = (char*)d_ws;
  size_t off = 0;
  u16* wf_bf = (u16*)(ws + off);   off += (size_t)H2DIM * H2DIM * 2;   // 42.5 MB
  u16* wiou_bf = (u16*)(ws + off); off += (size_t)H3DIM * H2DIM * 2;   // 63.7 MB
  float* iou_ws = (float*)(ws + off); off += (size_t)NNODES * H3DIM * 4;
  float* h_ws = (float*)(ws + off);   off += (size_t)NNODES * HDIM * 4;
  float* c_ws = (float*)(ws + off);   off += (size_t)NNODES * HDIM * 4;
  u16* hcat = (u16*)(ws + off);       off += (size_t)128 * H2DIM * 2;
  float* f_ws = (float*)(ws + off);   off += (size_t)128 * H2DIM * 4;
  float* msg_ws = (float*)(ws + off); off += (size_t)128 * H3DIM * 4;

  // weights -> bf16 (L2-resident afterwards: 106 MB < 192 MB L2)
  cvt_bf16_kernel<<<2048, 256, 0, stream>>>(Wf, wf_bf, (long)H2DIM * H2DIM);
  cvt_bf16_kernel<<<2048, 256, 0, stream>>>(Wiou, wiou_bf, (long)H3DIM * H2DIM);

  // encoder -> iou (255 x 6912)
  EncArgs ea;
  for (int i = 0; i < 12; ++i) ea.w[i] = (const float*)d_in[3 + i];
  ea.out_w = (const float*)d_in[15];
  ea.out_b = (const float*)d_in[16];
  ea.x = x;
  ea.iou = iou_ws;
  encoder_kernel<<<NNODES, 256, (13824 + 27648) * sizeof(float), stream>>>(ea);

  // leaves: apply_node(iou[127:], c_init[127:])
  apply_node_kernel<<<gblocks((long)128 * HDIM), 256, 0, stream>>>(
      iou_ws, c_init, h_ws, c_ws, b_iou, 127, (long)128 * HDIM);

  // upward pass
  for (int l = 6; l >= 0; --l) {
    int lo = (1 << l) - 1, n = 1 << l;
    int npad = (n + 15) & ~15;
    prep_up_kernel<<<gblocks((long)npad * H2DIM), 256, 0, stream>>>(h_ws, hcat, lo, n, npad);
    dim3 gf(H2DIM / 256, npad / 16);
    gemm_bf16_nt<<<gf, 256, 0, stream>>>(hcat, wf_bf, bf, f_ws, H2DIM, H2DIM, 1);
    dim3 gi(H3DIM / 256, npad / 16);
    gemm_bf16_nt<<<gi, 256, 0, stream>>>(hcat, wiou_bf, nullptr, msg_ws, H3DIM, H2DIM, 0);
    reduce_apply_kernel<<<n, 256, 0, stream>>>(iou_ws, h_ws, c_ws, msg_ws, f_ws,
                                               b_iou, s_iou, s_c, lo, 0);
  }

  // iou = relu(iou)
  relu_kernel<<<gblocks((long)NNODES * H3DIM), 256, 0, stream>>>(iou_ws, (long)NNODES * H3DIM);

  // root
  apply_node_kernel<<<gblocks((long)HDIM), 256, 0, stream>>>(
      iou_ws, c_ws, h_ws, c_ws, b_iou, 0, (long)HDIM);

  // downward pass
  for (int l = 1; l <= 7; ++l) {
    int lo = (1 << l) - 1, n = 1 << l;
    int npad = (n + 15) & ~15;
    prep_down_kernel<<<gblocks((long)npad * H2DIM), 256, 0, stream>>>(h_ws, hcat, lo, n, npad);
    dim3 gf(H2DIM / 256, npad / 16);
    gemm_bf16_nt<<<gf, 256, 0, stream>>>(hcat, wf_bf, bf, f_ws, H2DIM, H2DIM, 1);
    dim3 gi(H3DIM / 256, npad / 16);
    gemm_bf16_nt<<<gi, 256, 0, stream>>>(hcat, wiou_bf, nullptr, msg_ws, H3DIM, H2DIM, 0);
    reduce_apply_kernel<<<n, 256, 0, stream>>>(iou_ws, h_ws, c_ws, msg_ws, f_ws,
                                               b_iou, s_iou, s_c, lo, 1);
  }

  // decoder -> d_out (255 x 256, f32)
  DecArgs da;
  for (int i = 0; i < 12; ++i) da.w[i] = (const float*)d_in[17 + i];
  da.out_w = (const float*)d_in[29];
  da.out_b = (const float*)d_in[30];
  da.h = h_ws;
  da.out = (float*)d_out;
  decoder_kernel<<<NNODES, 256, 0, stream>>>(da);
}
